// LIFNetwork_87299505258541
// MI455X (gfx1250) — matrix operand order repaired and checked by reading, not compiled, not run on hardware
//
#include <hip/hip_runtime.h>

#define BATCH   128
#define NN      4096
#define TSTEPS  200
#define KCHUNK  256
#define NCHUNKS (NN / KCHUNK)          // 16
#define MROWS   32                     // batch rows per workgroup (2 M-tiles)
#define LDSROW  (KCHUNK + 8)           // 264 f16 -> 528B row stride (TDM pads 16B/row)

#define EXP_TAU 0.60653065971263342f   // exp(-dt/tau)
#define DTTAU   0.5f                   // dt/tau
#define EXP_SYN 0.60653065971263342f
#define DTSYN   0.5f

typedef __attribute__((ext_vector_type(16))) _Float16     v16h;
typedef __attribute__((ext_vector_type(8)))  _Float16     v8h;
typedef __attribute__((ext_vector_type(8)))  float        v8f;
typedef __attribute__((ext_vector_type(4)))  unsigned int u32x4;
typedef __attribute__((ext_vector_type(8)))  int          i32x8;
typedef __attribute__((ext_vector_type(4)))  int          i32x4;

// ---------------- W transpose + f32 -> f16 convert: Wt[n][k] = W[k][n] -------
__global__ void wt_convert(const float* __restrict__ W, _Float16* __restrict__ Wt) {
    __shared__ float tile[32][33];
    const int n0 = blockIdx.x * 32, k0 = blockIdx.y * 32;
    const int tx = threadIdx.x, ty = threadIdx.y;     // 32 x 8
#pragma unroll
    for (int i = 0; i < 4; ++i)
        tile[ty + i * 8][tx] = W[(size_t)(k0 + ty + i * 8) * NN + (n0 + tx)];
    __syncthreads();
#pragma unroll
    for (int i = 0; i < 4; ++i)
        Wt[(size_t)(n0 + ty + i * 8) * NN + (k0 + tx)] = (_Float16)tile[tx][ty + i * 8];
}

// ---------------- state init ------------------------------------------------
__global__ void init_state(const float* __restrict__ rec0, float* volts, float* rec,
                           float* sum_s, float* sum_v, _Float16* spk0) {
    const size_t i = (size_t)blockIdx.x * blockDim.x + threadIdx.x;
    volts[i] = 0.0f;
    rec[i]   = rec0[i];
    sum_s[i] = 0.0f;
    sum_v[i] = 0.0f;
    spk0[i]  = (_Float16)0.0f;
}

// ---------------- TDM: DMA a 32 x KCHUNK f16 tile (row stride NN) into LDS ---
// D# per CDNA5 ISA. Pads +16B after every 512B row so the
// LDS layout lands on the bank-staggered LDSROW=264 pitch. Issue from ONE wave.
__device__ __forceinline__ void tdm_load_tile(const _Float16* gsrc, unsigned lds_off) {
    const unsigned long long ga = (unsigned long long)(uintptr_t)gsrc;
    u32x4 g0;
    g0[0] = 1u;                                              // count=1, user D#
    g0[1] = lds_off;                                         // lds_addr (bytes)
    g0[2] = (unsigned)(ga & 0xffffffffull);                  // global_addr[31:0]
    g0[3] = (unsigned)((ga >> 32) & 0x1ffffffull)            // global_addr[56:32]
          | (2u << 30);                                      // type=2 ("image")
    i32x8 g1;
    g1[0] = (int)((1u << 16)      // data_size = 1 -> 2-byte elements
          |       (1u << 20)      // pad_enable
          |       (6u << 22)      // pad_interval: 128 DWORDs (= one 512B row)
          |       (3u << 25));    // pad_amount: 4 DWORDs (= 16B)
    g1[1] = (int)(((unsigned)NN & 0xffffu) << 16);           // tensor_dim0 = 4096 (lo16)
    g1[2] = (int)(((unsigned)BATCH & 0xffffu) << 16);        // dim0 hi=0, tensor_dim1=128
    g1[3] = (int)(((unsigned)KCHUNK & 0xffffu) << 16);       // dim1 hi=0, tile_dim0=256
    g1[4] = MROWS;                                           // tile_dim1=32, tile_dim2=0
    g1[5] = NN;                                              // tensor_dim0_stride lo32
    g1[6] = 0;                                               // stride0 hi, dim1_stride lo
    g1[7] = 0;
    const i32x4 gz4 = {0, 0, 0, 0};                          // 2D tile: groups 2/3 unused
    const i32x8 gz8 = {0, 0, 0, 0, 0, 0, 0, 0};
    __builtin_amdgcn_tensor_load_to_lds(g0, g1, gz4, gz4, gz8, 0);
}

// ---------------- one LIF timestep: GEMM + dynamics + recording -------------
__global__ void __launch_bounds__(256)
lif_step(const float* __restrict__ ff, const _Float16* __restrict__ Wt,
         const _Float16* __restrict__ spk_in, _Float16* __restrict__ spk_out,
         float* __restrict__ volts, float* __restrict__ rec,
         float* __restrict__ sum_s, float* __restrict__ sum_v,
         float* __restrict__ out, int t) {
    __shared__ __attribute__((aligned(16))) _Float16 As[2][MROWS * LDSROW];

    const int tid   = threadIdx.x;
    const int wave  = tid >> 5;
    const int lane  = tid & 31;
    const int lrow  = lane & 15;        // M row (A) / N col (B,C)
    const int lhalf = lane >> 4;        // K-half selector
    const int m0    = blockIdx.y * MROWS;              // 32 batch rows
    const int n0    = blockIdx.x * 128 + wave * 16;    // this wave's 16 columns

    v8f acc0 = {}, acc1 = {};           // two M-tiles, one N-tile (B reused 2x)

    // B fragment base (Wt is [n][k], contiguous along k)
    const _Float16* B0 = Wt + (size_t)(n0 + lrow) * NN + (size_t)lhalf * 8;
    const _Float16* spk_blk = spk_in + (size_t)m0 * NN;
    const unsigned lds0 = (unsigned)(unsigned long long)(const void*)&As[0][0];

    auto loadB = [&](v16h& b, int k) {
        const _Float16* bp = B0 + k;
        *(v8h*)&b       = *(const v8h*)(bp);
        *((v8h*)&b + 1) = *(const v8h*)(bp + 16);
    };
    auto loadA = [&](v16h& a0, v16h& a1, const _Float16* Ab, int kk) {
        const _Float16* ap0 = Ab + lrow * LDSROW + kk + lhalf * 8;
        const _Float16* ap1 = ap0 + 16 * LDSROW;
        *(v8h*)&a0       = *(const v8h*)(ap0);
        *((v8h*)&a0 + 1) = *(const v8h*)(ap0 + 16);
        *(v8h*)&a1       = *(const v8h*)(ap1);
        *((v8h*)&a1 + 1) = *(const v8h*)(ap1 + 16);
    };

    if (wave == 0) tdm_load_tile(spk_blk, lds0);            // stage chunk 0

    for (int c = 0; c < NCHUNKS; ++c) {
        const int kbase = c * KCHUNK;
        v16h bc, a0c, a1c;
        loadB(bc, kbase);                                    // in flight during barrier
        if (wave == 0) __builtin_amdgcn_s_wait_tensorcnt(0); // DMA for buf (c&1) done
        __syncthreads();                                     // publish LDS to all waves
        if (c + 1 < NCHUNKS) {
            if (wave == 0)                                   // DMA next chunk, overlapped
                tdm_load_tile(spk_blk + (c + 1) * KCHUNK,
                              lds0 + ((c + 1) & 1) * (unsigned)(MROWS * LDSROW * 2));
            __builtin_prefetch(B0 + (size_t)(c + 1) * KCHUNK, 0, 1);
        }
        const _Float16* Ab = &As[c & 1][0];
        loadA(a0c, a1c, Ab, 0);
        // distance-1 software pipeline: next fragments stay in flight across WMMAs
#pragma unroll
        for (int kk = 0; kk < KCHUNK; kk += 32) {
            v16h bn, a0n, a1n;
            const bool more = (kk + 32 < KCHUNK);            // compile-time after unroll
            if (more) {
                loadB(bn, kbase + kk + 32);
                loadA(a0n, a1n, Ab, kk + 32);
            }
            acc0 = __builtin_amdgcn_wmma_f32_16x16x32_f16(false, a0c, false, bc,
                                                          (short)0, acc0, false, false);
            acc1 = __builtin_amdgcn_wmma_f32_16x16x32_f16(false, a1c, false, bc,
                                                          (short)0, acc1, false, false);
            if (more) { bc = bn; a0c = a0n; a1c = a1n; }
        }
    }

    // ---------------- LIF dynamics + windowed recording ----------------------
    const bool in_win   = (t >= 80) && (t <= 180);
    const bool rec_step = (t >= 100) && (t % 20 == 0);
    const int  w        = (t - 100) / 20;
    const size_t OUT_PLANE = (size_t)BATCH * 5 * NN;
    const int n = n0 + lrow;

#pragma unroll
    for (int j = 0; j < 2; ++j) {                      // j = M-tile
#pragma unroll
        for (int r = 0; r < 8; ++r) {
            const int b = m0 + j * 16 + lhalf * 8 + r; // C layout: VGPR r -> row r / r+8
            const size_t idx = (size_t)b * NN + n;
            const float h  = (j == 0) ? acc0[r] : acc1[r];
            const float rc = rec[idx] * EXP_SYN + h * DTSYN;
            const float f  = ff[((size_t)b * TSTEPS + t) * NN + n];
            float v = volts[idx] * EXP_TAU + DTTAU * (f + rc);
            const float s = (v >= 1.0f) ? 1.0f : 0.0f;
            v = (v >= 1.0f) ? 0.0f : v;
            rec[idx]     = rc;
            volts[idx]   = v;
            spk_out[idx] = (_Float16)s;
            if (in_win) {
                float ss = sum_s[idx] + s;
                float sv = sum_v[idx] + v;
                if (rec_step) {
                    const size_t oidx = ((size_t)b * 5 + w) * NN + n;
                    out[oidx]                 = ss * (1.0f / 20.0f);  // rates
                    out[oidx + OUT_PLANE]     = sv * (1.0f / 20.0f);  // volts_avg
                    out[oidx + 2 * OUT_PLANE] = s;                    // spikes_rec
                    ss = 0.0f; sv = 0.0f;
                }
                sum_s[idx] = ss;
                sum_v[idx] = sv;
            }
        }
    }
}

// ---------------- host-side orchestration -----------------------------------
extern "C" void kernel_launch(void* const* d_in, const int* in_sizes, int n_in,
                              void* d_out, int out_size, void* d_ws, size_t ws_size,
                              hipStream_t stream) {
    const float* ff   = (const float*)d_in[0];   // [128, 200, 4096] f32
    const float* W    = (const float*)d_in[1];   // [4096, 4096] f32 (Wab_T)
    const float* rec0 = (const float*)d_in[2];   // [128, 4096] f32
    float* out = (float*)d_out;

    char* ws = (char*)d_ws;
    size_t o = 0;
    _Float16* Wt   = (_Float16*)(ws + o); o += (size_t)NN * NN * 2;        // 32 MB
    _Float16* spkA = (_Float16*)(ws + o); o += (size_t)BATCH * NN * 2;     //  1 MB
    _Float16* spkB = (_Float16*)(ws + o); o += (size_t)BATCH * NN * 2;     //  1 MB
    float* volts = (float*)(ws + o); o += (size_t)BATCH * NN * 4;
    float* rec   = (float*)(ws + o); o += (size_t)BATCH * NN * 4;
    float* sum_s = (float*)(ws + o); o += (size_t)BATCH * NN * 4;
    float* sum_v = (float*)(ws + o); o += (size_t)BATCH * NN * 4;

    wt_convert<<<dim3(NN / 32, NN / 32), dim3(32, 8), 0, stream>>>(W, Wt);
    init_state<<<(BATCH * NN) / 256, 256, 0, stream>>>(rec0, volts, rec, sum_s, sum_v, spkA);

    // outputs depend only on t <= 180
    for (int t = 0; t <= 180; ++t) {
        const _Float16* sin = (t & 1) ? spkB : spkA;
        _Float16*       sot = (t & 1) ? spkA : spkB;
        lif_step<<<dim3(32, 4), 256, 0, stream>>>(ff, Wt, sin, sot,
                                                  volts, rec, sum_s, sum_v, out, t);
    }
}
